// GLSTM_24369644437791
// MI455X (gfx1250) — matrix operand order repaired and checked by reading, not compiled, not run on hardware
//
#include <hip/hip_runtime.h>
#include <hip/hip_bf16.h>

// ---------- CDNA5 WMMA types ----------
typedef __bf16  v16bf __attribute__((ext_vector_type(16)));
typedef __bf16  v8bf  __attribute__((ext_vector_type(8)));
typedef float   v8f   __attribute__((ext_vector_type(8)));

__device__ __forceinline__ v8f wmma_bf16(v16bf a, v16bf b, v8f c) {
    // D = A(16x32) * B(32x16) + C, f32 accumulate
    return __builtin_amdgcn_wmma_f32_16x16x32_bf16(
        /*neg_a=*/false, a, /*neg_b=*/false, b,
        /*c_mod=*/(short)0, c, /*reuse_a=*/false, /*reuse_b=*/false);
}

// A fragment: 16x32 bf16 tile, row-major [m][k], leading dim ld (multiple of 8).
// Lane layout (ISA 7.12.2): M = lane&15; elements 0..7 <-> K = half*8+0..7,
// elements 8..15 <-> K = 16+half*8+0..7  => two contiguous 16-byte runs.
__device__ __forceinline__ v16bf load_a_frag(const __bf16* base, int ld, int lane) {
    const int half = (lane >> 4) & 1;
    const int m    = lane & 15;
    const __bf16* p = base + m * ld + half * 8;
    v16bf a;
#pragma unroll
    for (int e = 0; e < 8; ++e) { a[e] = p[e]; a[8 + e] = p[16 + e]; }
    return a;
}

// B fragment from N-major staging Bt[n][k] (ld multiple of 8).
// Lane layout: N = lane&15; elements 0..7 <-> K = half*8+0..7,
// elements 8..15 <-> K = 16+half*8+0..7  => two contiguous 16-byte runs.
__device__ __forceinline__ v16bf load_bt_frag(const __bf16* base, int ld, int lane) {
    const int half = (lane >> 4) & 1;
    const int n    = lane & 15;
    const __bf16* p = base + n * ld + half * 8;
    v16bf b;
#pragma unroll
    for (int e = 0; e < 8; ++e) { b[e] = p[e]; b[8 + e] = p[16 + e]; }
    return b;
}

// ---------- problem constants ----------
#define BATCH 4
#define HH 256
#define WW 256
#define QD 64
#define VD 256
#define NH 8
#define NWIN 4096              // 4 * 32 * 32
#define PIX 262144             // B*H*W
#define PLANE 65536            // H*W

__device__ __forceinline__ int win_of(int b, int h, int w) {
    return (b * 32 + (h >> 3)) * 32 + (w >> 3);
}
__device__ __forceinline__ int n_of(int h, int w) {
    return ((h & 7) << 3) + (w & 7);
}

// =====================================================================
// K1: qp = Wq*q ; kp = Wk*qp (fused). 128-pixel tile per block.
// Output layout: [win][head][n=64][d=8] bf16
// =====================================================================
__global__ void k1_proj_qk(const float* __restrict__ q,
                           const float* __restrict__ Wq,
                           const float* __restrict__ Wk,
                           __bf16* __restrict__ qp_win,
                           __bf16* __restrict__ kp_win) {
    __shared__ __align__(16) __bf16 sWq[64 * 72];     // A: [m][k], ld 72
    __shared__ __align__(16) __bf16 sWk[64 * 72];
    __shared__ __align__(16) __bf16 sQt[128 * 72];    // Bt: [j][c], ld 72
    __shared__ __align__(16) __bf16 sQPt[128 * 72];   // Bt: [j][c], ld 72

    const int tid  = threadIdx.x;
    const int wave = tid >> 5;
    const int lane = tid & 31;

    for (int i = tid; i < 64 * 64; i += 256) {
        int m = i >> 6, k = i & 63;
        sWq[m * 72 + k] = (__bf16)Wq[i];
        sWk[m * 72 + k] = (__bf16)Wk[i];
    }

    const long pix0 = (long)blockIdx.x * 128;
    const int  b    = (int)(pix0 >> 16);
    const int  rem  = (int)(pix0 & 65535);
    const int  h    = rem >> 8;
    const int  w0   = rem & 255;       // 0 or 128

    const float* qb = q + ((long)b * QD) * PLANE + h * WW + w0;
    for (int i = tid; i < 64 * 128; i += 256) {
        int c = i >> 7, j = i & 127;                  // contiguous j per run: coalesced global
        sQt[j * 72 + c] = (__bf16)qb[(long)c * PLANE + j];
    }
    __syncthreads();

    const int jt   = wave;             // wave owns pixel columns jt*16..+15
    const int half = lane >> 4;
    const int ncol = lane & 15;
    const int j    = jt * 16 + ncol;
    const int w    = w0 + j;
    const int win  = win_of(b, h, w);
    const int nn   = n_of(h, w);

    // ---- qp = Wq * q ----
#pragma unroll
    for (int mt = 0; mt < 4; ++mt) {
        v8f acc = {};
#pragma unroll
        for (int kt = 0; kt < 2; ++kt) {
            v16bf a  = load_a_frag(sWq + (mt * 16) * 72 + kt * 32, 72, lane);
            v16bf bb = load_bt_frag(sQt + (jt * 16) * 72 + kt * 32, 72, lane);
            acc = wmma_bf16(a, bb, acc);
        }
        // acc[r] -> channel c = mt*16 + 8*half + r : 8 consecutive channels = one head run
        int c0 = mt * 16 + 8 * half;
        v8bf pk;
#pragma unroll
        for (int r = 0; r < 8; ++r) pk[r] = (__bf16)acc[r];
        *(v8bf*)(&sQPt[j * 72 + c0]) = pk;                       // 16B LDS store
        long gbase = (((long)win * NH + (c0 >> 3)) * 64 + nn) * 8;
        *(v8bf*)(qp_win + gbase) = pk;                           // 16B global store
    }
    __syncthreads();

    // ---- kp = Wk * qp ----
#pragma unroll
    for (int mt = 0; mt < 4; ++mt) {
        v8f acc = {};
#pragma unroll
        for (int kt = 0; kt < 2; ++kt) {
            v16bf a  = load_a_frag(sWk + (mt * 16) * 72 + kt * 32, 72, lane);
            v16bf bb = load_bt_frag(sQPt + (jt * 16) * 72 + kt * 32, 72, lane);
            acc = wmma_bf16(a, bb, acc);
        }
        int c0 = mt * 16 + 8 * half;
        v8bf pk;
#pragma unroll
        for (int r = 0; r < 8; ++r) pk[r] = (__bf16)acc[r];
        long gbase = (((long)win * NH + (c0 >> 3)) * 64 + nn) * 8;
        *(v8bf*)(kp_win + gbase) = pk;
    }
}

// =====================================================================
// K2: vp = Wv*v (256x256, K sliced by 32). 64-pixel tile per block.
// Output layout: [win][head][n=64][dv=32] bf16
// =====================================================================
__global__ void k2_proj_v(const float* __restrict__ v,
                          const float* __restrict__ Wv,
                          __bf16* __restrict__ vp_win) {
    __shared__ __align__(16) __bf16 sW[256 * 40];     // A slice: [m][k], ld 40
    __shared__ __align__(16) __bf16 sVt[64 * 40];     // Bt slice: [j][k], ld 40

    const int tid  = threadIdx.x;
    const int wave = tid >> 5;
    const int lane = tid & 31;

    const long pix0 = (long)blockIdx.x * 64;
    const int  b    = (int)(pix0 >> 16);
    const int  rem  = (int)(pix0 & 65535);
    const int  h    = rem >> 8;
    const int  w0   = rem & 255;

    const float* vb = v + ((long)b * VD) * PLANE + h * WW + w0;

    v8f acc[8] = {};
    for (int ks = 0; ks < 8; ++ks) {
        for (int i = tid; i < 256 * 32; i += 256) {
            int m = i >> 5, k = i & 31;
            sW[m * 40 + k] = (__bf16)Wv[(long)m * 256 + ks * 32 + k];
        }
        for (int i = tid; i < 32 * 64; i += 256) {
            int k = i >> 6, jj = i & 63;
            sVt[jj * 40 + k] = (__bf16)vb[(long)(ks * 32 + k) * PLANE + jj];
        }
        __syncthreads();
#pragma unroll
        for (int t = 0; t < 8; ++t) {
            int id = wave * 8 + t;
            int mt = id >> 2, nt = id & 3;
            v16bf a  = load_a_frag(sW + (mt * 16) * 40, 40, lane);
            v16bf bb = load_bt_frag(sVt + (nt * 16) * 40, 40, lane);
            acc[t] = wmma_bf16(a, bb, acc[t]);
        }
        __syncthreads();
    }

    const int half = lane >> 4;
    const int ncol = lane & 15;
#pragma unroll
    for (int t = 0; t < 8; ++t) {
        int id = wave * 8 + t;
        int mt = id >> 2, nt = id & 3;
        int c0 = mt * 16 + 8 * half;          // 8 consecutive channels, same head
        int jj = nt * 16 + ncol;
        int w  = w0 + jj;
        int win = win_of(b, h, w);
        int nn  = n_of(h, w);
        v8bf pk;
#pragma unroll
        for (int r = 0; r < 8; ++r) pk[r] = (__bf16)acc[t][r];
        long gbase = ((((long)win * NH + (c0 >> 5)) * 64 + nn) * 32) + (c0 & 31);
        *(v8bf*)(vp_win + gbase) = pk;
    }
}

// =====================================================================
// K3: window attention. One wave (32 threads) = one (window, head).
// grid = NWIN * NH. dots (K=8 zero-padded to 32) -> softmax+bias -> attn*v.
// =====================================================================
__global__ void k3_attn(const __bf16* __restrict__ qp_win,
                        const __bf16* __restrict__ kp_win,
                        const __bf16* __restrict__ vp_win,
                        const float* __restrict__ bias_table,
                        const int* __restrict__ rel_index,
                        float* __restrict__ attn_out) {
    __shared__ __align__(16) __bf16 sQ [64 * 40];   // A: [n][d pad 32->40], zero padded
    __shared__ __align__(16) __bf16 sKt[64 * 40];   // Bt: [n][d pad], zero padded
    __shared__ __align__(16) __bf16 sVt[32 * 72];   // Bt for out: [dv][n=64 pad 72]
    __shared__ __align__(16) __bf16 sA [64 * 72];   // A for out: [i][j pad 72]
    __shared__            float  sD [64 * 65];      // dots scratch, ld 65

    const int lane = threadIdx.x & 31;
    const int win  = blockIdx.x >> 3;
    const int head = blockIdx.x & 7;

    const long base8  = (((long)win * NH + head) * 64) * 8;
    const long base32 = (((long)win * NH + head) * 64) * 32;

    for (int i = lane; i < 64 * 40; i += 32) { sQ[i] = (__bf16)0.0f; sKt[i] = (__bf16)0.0f; }
    __syncthreads();
    for (int i = lane; i < 512; i += 32) {
        int n = i >> 3, d = i & 7;
        sQ [n * 40 + d] = qp_win[base8 + i];
        sKt[n * 40 + d] = kp_win[base8 + i];
    }
    for (int i = lane; i < 2048; i += 32) {
        int n = i >> 5, dv = i & 31;
        sVt[dv * 72 + n] = vp_win[base32 + i];      // transpose staging
    }
    __syncthreads();

    const float scale = 0.35355339059327373f;   // (QD/NH)^-0.5
    const int half = lane >> 4;
    const int m    = lane & 15;

    // ---- dots = scale * q @ k^T : 16 tiles of 16x16, K = 32 (zero padded) ----
    for (int mt = 0; mt < 4; ++mt) {
        for (int nt = 0; nt < 4; ++nt) {
            v16bf a  = load_a_frag(sQ + (mt * 16) * 40, 40, lane);
            v16bf bb = load_bt_frag(sKt + (nt * 16) * 40, 40, lane);
            v8f acc = {};
            acc = wmma_bf16(a, bb, acc);
#pragma unroll
            for (int r = 0; r < 8; ++r) {
                int i = mt * 16 + 8 * half + r;
                int jj = nt * 16 + m;
                sD[i * 65 + jj] = acc[r] * scale;
            }
        }
    }
    __syncthreads();

    // ---- softmax(row + bias): each lane owns 2 rows ----
    for (int rr = 0; rr < 2; ++rr) {
        int i = lane * 2 + rr;
        float*     row = &sD[i * 65];
        const int* ri  = rel_index + i * 64;
        float mx = -3.0e38f;
        for (int jj = 0; jj < 64; ++jj) {
            float t = row[jj] + bias_table[ri[jj] * NH + head];
            row[jj] = t;
            mx = fmaxf(mx, t);
        }
        float s = 0.f;
        for (int jj = 0; jj < 64; ++jj) {
            float e = __expf(row[jj] - mx);
            row[jj] = e;
            s += e;
        }
        float inv = 1.0f / s;
        for (int jj = 0; jj < 64; ++jj)
            sA[i * 72 + jj] = (__bf16)(row[jj] * inv);
    }
    __syncthreads();

    // ---- out = attn(64x64) @ v(64x32) ----
    const int b  = win >> 10;
    const int rw = win & 1023;
    const int wh = rw >> 5;
    const int wx = rw & 31;
    for (int mt = 0; mt < 4; ++mt) {
        for (int nt = 0; nt < 2; ++nt) {
            v8f acc = {};
#pragma unroll
            for (int kt = 0; kt < 2; ++kt) {
                v16bf a  = load_a_frag(sA + (mt * 16) * 72 + kt * 32, 72, lane);
                v16bf bb = load_bt_frag(sVt + (nt * 16) * 72 + kt * 32, 72, lane);
                acc = wmma_bf16(a, bb, acc);
            }
            // acc[r] -> window n index i0+r, i0 = mt*16+8*half: 8 consecutive px
            int i0 = mt * 16 + 8 * half;
            int py = wh * 8 + (i0 >> 3);
            int c  = head * 32 + nt * 16 + m;
            float* dst = attn_out + ((long)b * VD + c) * PLANE + py * WW + wx * 8;
            *(v8f*)dst = acc;                       // 32B contiguous f32 store
        }
    }
}

// =====================================================================
// K4: reflect-pad(+1) + zero-pad(3) 8x8 depthwise conv + eval BN -> bf16
// One thread per output element; wave = one channel's contiguous 32 px.
// =====================================================================
__global__ void k4_dwbn(const float* __restrict__ attn_out,
                        const float* __restrict__ dwk,
                        const float* __restrict__ gamma,
                        const float* __restrict__ beta,
                        __bf16* __restrict__ dwout) {
    const int lane = threadIdx.x & 31;
    const int wv   = threadIdx.x >> 5;
    const int bid  = blockIdx.x;        // [b(2)][h(8)][wt(3)][cg(5)]
    const int cg = bid & 31;
    const int wt = (bid >> 5) & 7;
    const int h  = (bid >> 8) & 255;
    const int b  = bid >> 16;
    const int c  = cg * 8 + wv;
    const int w  = wt * 32 + lane;

    const float* plane = attn_out + ((long)b * VD + c) * PLANE;
    float acc = 0.f;
#pragma unroll
    for (int ky = 0; ky < 8; ++ky) {
        int y = h - 3 + ky;
        if (y < 0 || y > 256) continue;     // zero pad
        int ys = (y == 256) ? 254 : y;      // reflect (exclude edge)
        const float* rowp = plane + ys * WW;
#pragma unroll
        for (int kx = 0; kx < 8; ++kx) {
            int x = w - 3 + kx;
            if (x < 0 || x > 256) continue;
            int xs = (x == 256) ? 254 : x;
            acc += dwk[c * 64 + ky * 8 + kx] * rowp[xs];
        }
    }
    const float gi = gamma[c] * rsqrtf(1.0f + 1e-5f);
    dwout[((long)b * VD + c) * PLANE + h * WW + w] = (__bf16)(acc * gi + beta[c]);
}

// =====================================================================
// K5: out = pw(256x256) @ dw  (same structure as K2, bf16 B from ws)
// =====================================================================
__global__ void k5_pw(const __bf16* __restrict__ dwout,
                      const float* __restrict__ pw,
                      float* __restrict__ out) {
    __shared__ __align__(16) __bf16 sW[256 * 40];
    __shared__ __align__(16) __bf16 sXt[64 * 40];

    const int tid  = threadIdx.x;
    const int wave = tid >> 5;
    const int lane = tid & 31;

    const long pix0 = (long)blockIdx.x * 64;
    const int  b    = (int)(pix0 >> 16);
    const int  rem  = (int)(pix0 & 65535);
    const int  h    = rem >> 8;
    const int  w0   = rem & 255;

    const __bf16* xb = dwout + ((long)b * VD) * PLANE + h * WW + w0;

    v8f acc[8] = {};
    for (int ks = 0; ks < 8; ++ks) {
        for (int i = tid; i < 256 * 32; i += 256) {
            int mrow = i >> 5, k = i & 31;
            sW[mrow * 40 + k] = (__bf16)pw[(long)mrow * 256 + ks * 32 + k];
        }
        for (int i = tid; i < 32 * 64; i += 256) {
            int k = i >> 6, jj = i & 63;
            sXt[jj * 40 + k] = xb[(long)(ks * 32 + k) * PLANE + jj];
        }
        __syncthreads();
#pragma unroll
        for (int t = 0; t < 8; ++t) {
            int id = wave * 8 + t;
            int mt = id >> 2, nt = id & 3;
            v16bf a  = load_a_frag(sW + (mt * 16) * 40, 40, lane);
            v16bf bb = load_bt_frag(sXt + (nt * 16) * 40, 40, lane);
            acc[t] = wmma_bf16(a, bb, acc[t]);
        }
        __syncthreads();
    }

    const int half = lane >> 4;
    const int ncol = lane & 15;
#pragma unroll
    for (int t = 0; t < 8; ++t) {
        int id = wave * 8 + t;
        int mt = id >> 2, nt = id & 3;
#pragma unroll
        for (int r = 0; r < 8; ++r) {
            int c = mt * 16 + 8 * half + r;
            int jj = nt * 16 + ncol;
            out[((long)b * VD + c) * PLANE + h * WW + (w0 + jj)] = acc[t][r];
        }
    }
}

// =====================================================================
extern "C" void kernel_launch(void* const* d_in, const int* in_sizes, int n_in,
                              void* d_out, int out_size, void* d_ws, size_t ws_size,
                              hipStream_t stream) {
    (void)in_sizes; (void)n_in; (void)out_size; (void)ws_size;
    const float* q          = (const float*)d_in[0];
    const float* v          = (const float*)d_in[1];
    const float* Wq         = (const float*)d_in[2];
    const float* Wk         = (const float*)d_in[3];
    const float* Wv         = (const float*)d_in[4];
    const float* bias_table = (const float*)d_in[5];
    const float* dw_kernel  = (const float*)d_in[6];
    const float* bn_gamma   = (const float*)d_in[7];
    const float* bn_beta    = (const float*)d_in[8];
    const float* pw_kernel  = (const float*)d_in[9];
    const int*   rel_index  = (const int*)d_in[10];
    float* out = (float*)d_out;

    // workspace layout (bytes): qp 33.5MB | kp 33.5MB | vp 134MB | attn 268MB
    char* ws = (char*)d_ws;
    __bf16* qp_win = (__bf16*)(ws);
    __bf16* kp_win = (__bf16*)(ws + (size_t)33554432);
    __bf16* vp_win = (__bf16*)(ws + (size_t)67108864);            // reused by K4 as dwout
    float*  attn_o = (float*) (ws + (size_t)201326592);
    __bf16* dwout  = vp_win;   // vp dead after K3; same element count (67.1M)

    k1_proj_qk<<<PIX / 128, 256, 0, stream>>>(q, Wq, Wk, qp_win, kp_win);
    k2_proj_v <<<PIX / 64, 256, 0, stream>>>(v, Wv, vp_win);
    k3_attn   <<<NWIN * NH, 32, 0, stream>>>(qp_win, kp_win, vp_win,
                                             bias_table, rel_index, attn_o);
    k4_dwbn   <<<BATCH * HH * 8 * 32, 256, 0, stream>>>(attn_o, dw_kernel,
                                                        bn_gamma, bn_beta, dwout);
    k5_pw     <<<PIX / 64, 256, 0, stream>>>(dwout, pw_kernel, out);
}
